// Decoder_36739150250340
// MI455X (gfx1250) — matrix operand order repaired and checked by reading, not compiled, not run on hardware
//
#include <hip/hip_runtime.h>
#include <hip/hip_bf16.h>

// ---------------------------------------------------------------------------
// MI455X (gfx1250) transformer block: all GEMMs via v_wmma_f32_16x16x32_bf16,
// staged through the CDNA5 async global->LDS path (ASYNCcnt-tracked).
// ---------------------------------------------------------------------------

typedef __attribute__((ext_vector_type(16))) __bf16 v16bf;
typedef __attribute__((ext_vector_type(8)))  float  v8f;

#define TM 128
#define TN 128
#define TK 64
#define LDSS 72  // LDS row stride in bf16 elems (144 B: 16B-aligned, conflict-free)

__device__ __forceinline__ v16bf frag_ld(const unsigned short* p0,
                                         const unsigned short* p1) {
  union { uint4 u[2]; v16bf v; } f;
  f.u[0] = *(const uint4*)p0;
  f.u[1] = *(const uint4*)p1;
  return f.v;
}

// Async 16B global -> LDS copy (per lane). GVS addressing: SGPR64 base +
// 32-bit per-lane byte offset. Tracked by ASYNCcnt.
__device__ __forceinline__ void async_ld16(unsigned lds_off, const void* base,
                                           unsigned goff) {
  asm volatile("global_load_async_to_lds_b128 %0, %1, %2"
               :: "v"(lds_off), "v"(goff), "s"(base) : "memory");
}
__device__ __forceinline__ void wait_async_le8() {
  asm volatile("s_wait_asynccnt 0x8" ::: "memory");
}
__device__ __forceinline__ void wait_async_0() {
  asm volatile("s_wait_asynccnt 0x0" ::: "memory");
}

// EPI: 0 = bf16 store, 1 = f32 store, 2 = bf16 ReLU store,
//      3 = bf16 store transposed within each 2048-row batch (v^T),
//      4 = no store; block-sum + atomicAdd into redOut (final sum(y)).
template <int EPI>
__global__ void __launch_bounds__(256)
gemm_nt(const __hip_bfloat16* __restrict__ A,   // M x K row-major (bf16)
        const __hip_bfloat16* __restrict__ B,   // N x K row-major (bf16) => C = A * B^T
        void* __restrict__ Cout,
        int M, int N, int K,
        float* __restrict__ redOut)
{
  __shared__ __align__(16) unsigned short sA[2][TM][LDSS];
  __shared__ __align__(16) unsigned short sB[2][TN][LDSS];
  __shared__ float red[256];

  const int tid  = threadIdx.x;
  const int lane = tid & 31;
  const int wv   = tid >> 5;     // 8 waves
  const int wm   = wv & 3;       // 4 waves down M (32 rows each)
  const int wn   = wv >> 2;      // 2 waves across N (64 cols each)
  const int hl   = lane >> 4;    // lane half
  const int ml   = lane & 15;
  const int bM   = blockIdx.y * TM;
  const int bN   = blockIdx.x * TN;

  // Staging split: 128 rows x 64 cols per tile; 256 threads x 4 chunks of 16B.
  const int cc8 = (tid & 7) * 8;   // chunk column (8 bf16 = 16B)
  const int rr8 = tid >> 3;        // base row 0..31 (+32/64/96)

  // Issue one K-tile (A and B) of async global->LDS copies: 8 ops per thread.
  auto issue_tile = [&](int buf, int k0) {
#pragma unroll
    for (int q = 0; q < 4; ++q) {
      const int row = rr8 + q * 32;
      const unsigned la = (unsigned)(size_t)&sA[buf][row][cc8];
      const unsigned lb = (unsigned)(size_t)&sB[buf][row][cc8];
      const unsigned ga = (unsigned)(((size_t)(bM + row) * K + k0 + cc8) * 2);
      const unsigned gb = (unsigned)(((size_t)(bN + row) * K + k0 + cc8) * 2);
      async_ld16(la, A, ga);
      async_ld16(lb, B, gb);
    }
  };

  const int KT = K / TK;
  issue_tile(0, 0);
  issue_tile(1, TK);

  v8f acc[2][4] = {};

  for (int kt = 0; kt < KT; ++kt) {
    if (kt + 1 < KT) wait_async_le8();   // this tile landed; next still in flight
    else             wait_async_0();
    __syncthreads();

    const int p = kt & 1;
#pragma unroll
    for (int s = 0; s < 2; ++s) {        // two 32-wide WMMA K-steps per tile
      const int kb = s * 32;
      // A fragment: lane-half 0 holds K {0..7,16..23}, half 1 {8..15,24..31}
      v16bf af[2];
#pragma unroll
      for (int i = 0; i < 2; ++i) {
        const int row = wm * 32 + i * 16 + ml;
        af[i] = frag_ld(&sA[p][row][kb + hl * 8], &sA[p][row][kb + 16 + hl * 8]);
      }
      // B fragment: lanes 0-15 col N=lane K {0..7,16..23}; lanes 16-31 same col K {8..15,24..31}
      v16bf bfr[4];
#pragma unroll
      for (int j = 0; j < 4; ++j) {
        const int col = wn * 64 + j * 16 + ml;
        bfr[j] = frag_ld(&sB[p][col][kb + hl * 8], &sB[p][col][kb + 16 + hl * 8]);
      }
#pragma unroll
      for (int i = 0; i < 2; ++i)
#pragma unroll
        for (int j = 0; j < 4; ++j)
          acc[i][j] = __builtin_amdgcn_wmma_f32_16x16x32_bf16(
              false, af[i], false, bfr[j], (short)0, acc[i][j], false, false);
    }

    __syncthreads();
    if (kt + 2 < KT) issue_tile(p, (kt + 2) * TK);
  }

  if (EPI == 4) {
    float local = 0.f;
#pragma unroll
    for (int i = 0; i < 2; ++i)
#pragma unroll
      for (int j = 0; j < 4; ++j)
#pragma unroll
        for (int e = 0; e < 8; ++e) local += acc[i][j][e];
    red[tid] = local;
    __syncthreads();
    for (int off = 128; off > 0; off >>= 1) {
      if (tid < off) red[tid] += red[tid + off];
      __syncthreads();
    }
    if (tid == 0) atomicAdd(redOut, red[0]);
    return;
  }

  // C layout: VGPR e, lanes 0-15 -> M = e, N = lane; lanes 16-31 -> M = 8+e
#pragma unroll
  for (int i = 0; i < 2; ++i) {
    const int mb = bM + wm * 32 + i * 16 + hl * 8;
#pragma unroll
    for (int j = 0; j < 4; ++j) {
      const int n = bN + wn * 64 + j * 16 + ml;
#pragma unroll
      for (int e = 0; e < 8; ++e) {
        const int m = mb + e;
        const float v = acc[i][j][e];
        if (EPI == 1) {
          ((float*)Cout)[(size_t)m * N + n] = v;
        } else if (EPI == 0) {
          ((__hip_bfloat16*)Cout)[(size_t)m * N + n] = __float2bfloat16(v);
        } else if (EPI == 2) {
          ((__hip_bfloat16*)Cout)[(size_t)m * N + n] =
              __float2bfloat16(fmaxf(v, 0.f));
        } else {  // EPI == 3: v^T per batch of 2048 rows: out[b][n][s]
          const int b = m >> 11, s = m & 2047;
          ((__hip_bfloat16*)Cout)[((size_t)b << 22) + ((size_t)n << 11) + s] =
              __float2bfloat16(v);
        }
      }
    }
  }
}

__global__ void __launch_bounds__(256)
softmax_rows(const float* __restrict__ S, __hip_bfloat16* __restrict__ P, int n)
{
  __shared__ float red[256];
  const int row = blockIdx.x;
  const float* s = S + (size_t)row * n;
  __hip_bfloat16* p = P + (size_t)row * n;
  const int tid = threadIdx.x;

  float m = -3.0e38f;
  for (int i = tid; i < n; i += 256) m = fmaxf(m, s[i]);
  red[tid] = m; __syncthreads();
  for (int off = 128; off > 0; off >>= 1) {
    if (tid < off) red[tid] = fmaxf(red[tid], red[tid + off]);
    __syncthreads();
  }
  m = red[0]; __syncthreads();

  float sum = 0.f;
  for (int i = tid; i < n; i += 256) sum += __expf(s[i] - m);
  red[tid] = sum; __syncthreads();
  for (int off = 128; off > 0; off >>= 1) {
    if (tid < off) red[tid] += red[tid + off];
    __syncthreads();
  }
  const float inv = 1.0f / red[0];
  for (int i = tid; i < n; i += 256)
    p[i] = __float2bfloat16(__expf(s[i] - m) * inv);
}

__global__ void __launch_bounds__(256)
cast_f32_to_bf16(const float* __restrict__ src,
                 __hip_bfloat16* __restrict__ dst, size_t n)
{
  size_t i = (size_t)blockIdx.x * blockDim.x + threadIdx.x;
  const size_t stride = (size_t)gridDim.x * blockDim.x;
  for (; i < n; i += stride) dst[i] = __float2bfloat16(src[i]);
}

__global__ void zero_f32(float* __restrict__ out, int n)
{
  int i = blockIdx.x * blockDim.x + threadIdx.x;
  if (i < n) out[i] = 0.f;
}

extern "C" void kernel_launch(void* const* d_in, const int* in_sizes, int n_in,
                              void* d_out, int out_size, void* d_ws, size_t ws_size,
                              hipStream_t stream) {
  const float* x  = (const float*)d_in[0];
  const float* Ws[6] = { (const float*)d_in[1], (const float*)d_in[2],
                         (const float*)d_in[3], (const float*)d_in[4],
                         (const float*)d_in[5], (const float*)d_in[6] };
  float* out = (float*)d_out;

  const size_t S_X = 8192ull * 2048ull;   // x-shaped activations (elems)
  const size_t S_W = 2048ull * 2048ull;   // one weight / one batch tile (elems)

  char* w = (char*)d_ws;
  size_t off = 0;
  auto take = [&](size_t bytes) -> void* {
    void* p = w + off;
    off += (bytes + 255) & ~(size_t)255;
    return p;
  };

  __hip_bfloat16* xb  = (__hip_bfloat16*)take(S_X * 2);
  __hip_bfloat16* wb[6];
  for (int i = 0; i < 6; ++i) wb[i] = (__hip_bfloat16*)take(S_W * 2);
  __hip_bfloat16* qb  = (__hip_bfloat16*)take(S_X * 2);
  __hip_bfloat16* kb  = (__hip_bfloat16*)take(S_X * 2);
  __hip_bfloat16* vtb = (__hip_bfloat16*)take(S_X * 2);   // v^T per batch
  float*          sc  = (float*)take(4ull * S_W * 4);     // attention logits f32
  __hip_bfloat16* pb  = (__hip_bfloat16*)take(S_X * 2);   // softmax probs
  __hip_bfloat16* atb = (__hip_bfloat16*)take(S_X * 2);   // score @ v
  __hip_bfloat16* ob  = (__hip_bfloat16*)take(S_X * 2);   // attn @ Wo^T
  __hip_bfloat16* hb  = (__hip_bfloat16*)take(S_X * 2);   // relu(o @ W1^T)
  (void)ws_size; (void)in_sizes; (void)n_in;

  zero_f32<<<1, 32, 0, stream>>>(out, out_size);
  cast_f32_to_bf16<<<2048, 256, 0, stream>>>(x, xb, S_X);
  for (int i = 0; i < 6; ++i)
    cast_f32_to_bf16<<<512, 256, 0, stream>>>(Ws[i], wb[i], S_W);

  const dim3 blk(256);
  const dim3 gFull(2048 / TN, 8192 / TM);  // (16, 64)
  const dim3 gSq(2048 / TN, 2048 / TM);    // (16, 16)

  // q/k/v projections (v written transposed per batch)
  gemm_nt<0><<<gFull, blk, 0, stream>>>(xb, wb[0], qb,  8192, 2048, 2048, nullptr);
  gemm_nt<0><<<gFull, blk, 0, stream>>>(xb, wb[1], kb,  8192, 2048, 2048, nullptr);
  gemm_nt<3><<<gFull, blk, 0, stream>>>(xb, wb[2], vtb, 8192, 2048, 2048, nullptr);

  // logits = q @ k^T per batch (f32)
  for (int b = 0; b < 4; ++b)
    gemm_nt<1><<<gSq, blk, 0, stream>>>(qb + (size_t)b * S_W, kb + (size_t)b * S_W,
                                        sc + (size_t)b * S_W, 2048, 2048, 2048, nullptr);

  softmax_rows<<<8192, 256, 0, stream>>>(sc, pb, 2048);

  // attn = P @ v  ==  P @ (v^T)^T  -> NT GEMM per batch
  for (int b = 0; b < 4; ++b)
    gemm_nt<0><<<gSq, blk, 0, stream>>>(pb + (size_t)b * S_W, vtb + (size_t)b * S_W,
                                        atb + (size_t)b * S_W, 2048, 2048, 2048, nullptr);

  // output proj, FFN, and final reduction sum(y)
  gemm_nt<0><<<gFull, blk, 0, stream>>>(atb, wb[3], ob, 8192, 2048, 2048, nullptr);
  gemm_nt<2><<<gFull, blk, 0, stream>>>(ob,  wb[4], hb, 8192, 2048, 2048, nullptr);
  gemm_nt<4><<<gFull, blk, 0, stream>>>(hb,  wb[5], nullptr, 8192, 2048, 2048, out);
}
